// GCN_28991029248867
// MI455X (gfx1250) — matrix operand order repaired and checked by reading, not compiled, not run on hardware
//
#include <hip/hip_runtime.h>
#include <hip/hip_bf16.h>

// ---------- CDNA5 WMMA types ----------
typedef __bf16  v8bf  __attribute__((ext_vector_type(8)));
typedef __bf16  v16bf __attribute__((ext_vector_type(16)));
typedef float   v8f   __attribute__((ext_vector_type(8)));

union AFrag { v16bf v16; v8bf v8[2]; };

// ---------- elementwise f32 -> bf16 ----------
__global__ void cvt_f32_bf16(const float* __restrict__ src,
                             __bf16* __restrict__ dst, long n) {
    long stride = (long)gridDim.x * blockDim.x;
    for (long i = (long)blockIdx.x * blockDim.x + threadIdx.x; i < n; i += stride)
        dst[i] = (__bf16)src[i];
}

// ---------- pack W (K x Nreal, row-major, f32) into WMMA B-fragment order ----------
// Layout: Bp[((kt*NT + nt)*32 + lane)*16 + j] = W[kt*32 + (lane<16?0:16) + j][nt*16 + lane%16]
__global__ void pack_B(const float* __restrict__ W, __bf16* __restrict__ Bp,
                       int Kt, int NT, int Nreal) {
    int tid  = blockIdx.x * blockDim.x + threadIdx.x;
    int total = Kt * NT * 32;
    if (tid >= total) return;
    int lane = tid & 31;
    int tile = tid >> 5;
    int nt = tile % NT;
    int kt = tile / NT;
    int n  = nt * 16 + (lane & 15);
    int kbase = kt * 32 + ((lane < 16) ? 0 : 16);
    __bf16* out = Bp + (long)tid * 16;
    for (int j = 0; j < 16; ++j) {
        float v = (n < Nreal) ? W[(long)(kbase + j) * Nreal + n] : 0.0f;
        out[j] = (__bf16)v;
    }
}

// ---------- bf16 WMMA GEMM: C[M x Nreal] = A[M x K] * B(packed) ----------
// One wave computes one 16-row M-tile and NTW consecutive 16-col N-tiles.
template <int NTW>
__global__ void gemm_wmma_bf16(const __bf16* __restrict__ A,
                               const __bf16* __restrict__ Bp,
                               float* __restrict__ C,
                               int Mtiles, int Kt, int NTtot, int Nreal) {
    const int wave = threadIdx.x >> 5;
    const int lane = threadIdx.x & 31;
    const int mt   = blockIdx.x * (blockDim.x >> 5) + wave;
    if (mt >= Mtiles) return;
    const int ntBase = blockIdx.y * NTW;

    const int K = Kt * 32;
    const int m = mt * 16 + (lane & 15);
    const __bf16* Arow = A + (long)m * K;
    const int koffA = (lane < 16) ? 0 : 16;  // contiguous A-fragment layout (matches packed B)

    v8f acc[NTW];
    const v8f zero = {};
#pragma unroll
    for (int n = 0; n < NTW; ++n) acc[n] = zero;

    for (int kt = 0; kt < Kt; ++kt) {
        AFrag a;
        a.v8[0] = *(const v8bf*)(Arow + kt * 32 + koffA);       // K = c .. c+7
        a.v8[1] = *(const v8bf*)(Arow + kt * 32 + koffA + 8);   // K = c+8 .. c+15
#pragma unroll
        for (int n = 0; n < NTW; ++n) {
            v16bf b = *(const v16bf*)(Bp + (((long)kt * NTtot + (ntBase + n)) * 32 + lane) * 16);
            acc[n] = __builtin_amdgcn_wmma_f32_16x16x32_bf16(
                false, a.v16, false, b, (short)0, acc[n], false, false);
        }
    }

    // D layout: VGPR v -> row = v + 8*(lane>=16), col = lane%16
    const int rowBase = mt * 16 + ((lane >= 16) ? 8 : 0);
    const int col0 = lane & 15;
#pragma unroll
    for (int n = 0; n < NTW; ++n) {
        int col = (ntBase + n) * 16 + col0;
        if (col >= Nreal) continue;
#pragma unroll
        for (int v = 0; v < 8; ++v)
            C[(long)(rowBase + v) * Nreal + col] = acc[n][v];
    }
}

// ---------- SpMM via per-edge wave + f32 global atomics (L2-resident) ----------
__global__ void spmm_atomic(const int* __restrict__ erow, const int* __restrict__ ecol,
                            const float* __restrict__ ev,
                            const float* __restrict__ X, float* __restrict__ Y,
                            int E, int D) {
    int wid  = (blockIdx.x * blockDim.x + threadIdx.x) >> 5;
    int lane = threadIdx.x & 31;
    if (wid >= E) return;
    int r = erow[wid];
    int c = ecol[wid];
    float v = ev[wid];
    const float* xr = X + (long)c * D;
    float*       yr = Y + (long)r * D;
    for (int d = lane; d < D; d += 32)
        atomicAdd(&yr[d], v * xr[d]);
}

// ---------- epilogue: mid = relu(agg + b); also emit bf16 copy for layer 2 ----------
__global__ void bias_relu_store(const float* __restrict__ agg, const float* __restrict__ b,
                                float* __restrict__ outMid, __bf16* __restrict__ midBf,
                                long total, int D) {
    long stride = (long)gridDim.x * blockDim.x;
    for (long i = (long)blockIdx.x * blockDim.x + threadIdx.x; i < total; i += stride) {
        float v = agg[i] + b[i % D];
        v = fmaxf(v, 0.0f);
        outMid[i] = v;
        midBf[i]  = (__bf16)v;
    }
}

// ---------- epilogue: out = agg + b ----------
__global__ void bias_store(const float* __restrict__ agg, const float* __restrict__ b,
                           float* __restrict__ out, long total, int D) {
    long stride = (long)gridDim.x * blockDim.x;
    for (long i = (long)blockIdx.x * blockDim.x + threadIdx.x; i < total; i += stride)
        out[i] = agg[i] + b[i % D];
}

static inline size_t alignUp(size_t x, size_t a) { return (x + a - 1) & ~(a - 1); }

extern "C" void kernel_launch(void* const* d_in, const int* in_sizes, int n_in,
                              void* d_out, int out_size, void* d_ws, size_t ws_size,
                              hipStream_t stream) {
    const float* x    = (const float*)d_in[0];
    const int*   erow = (const int*)  d_in[1];
    const int*   ecol = (const int*)  d_in[2];
    const float* ev   = (const float*)d_in[3];
    const float* W1   = (const float*)d_in[4];
    const float* b1   = (const float*)d_in[5];
    const float* W2   = (const float*)d_in[6];
    const float* b2   = (const float*)d_in[7];

    const int E = in_sizes[1];
    const int H = in_sizes[5];          // 256
    const int C = in_sizes[7];          // 40
    const int F = in_sizes[4] / H;      // 512
    const int N = in_sizes[0] / F;      // 100000

    const int Kt1 = F / 32, NT1 = H / 16;              // 16, 16
    const int Cpad = ((C + 15) / 16) * 16;             // 48
    const int Kt2 = H / 32, NT2 = Cpad / 16;           // 8, 3
    const int Mtiles = (N + 15) / 16;                  // 6250

    // ---- workspace carve ----
    char* p = (char*)d_ws;
    auto carve = [&](size_t bytes) { char* r = p; p += alignUp(bytes, 256); return r; };
    __bf16* xBf   = (__bf16*)carve((size_t)N * F * sizeof(__bf16));
    __bf16* Bp1   = (__bf16*)carve((size_t)F * H * sizeof(__bf16));
    float*  xw1   = (float*) carve((size_t)N * H * sizeof(float));
    float*  agg1  = (float*) carve((size_t)N * H * sizeof(float));
    __bf16* midBf = (__bf16*)carve((size_t)N * H * sizeof(__bf16));
    __bf16* Bp2   = (__bf16*)carve((size_t)H * Cpad * sizeof(__bf16));
    float*  xw2   = (float*) carve((size_t)N * C * sizeof(float));
    float*  agg2  = (float*) carve((size_t)N * C * sizeof(float));
    (void)ws_size;

    float* outMid = (float*)d_out;                  // [N, H]
    float* outFin = (float*)d_out + (long)N * H;    // [N, C]

    // ---- zero accumulators (graph-capture legal) ----
    hipMemsetAsync(agg1, 0, (size_t)N * H * sizeof(float), stream);
    hipMemsetAsync(agg2, 0, (size_t)N * C * sizeof(float), stream);

    // ---- 1) x -> bf16 ----
    {
        long n = (long)N * F;
        int blocks = 2048;
        cvt_f32_bf16<<<blocks, 256, 0, stream>>>(x, xBf, n);
    }
    // ---- 2) pack weights into B-fragment layout ----
    {
        int t1 = Kt1 * NT1 * 32;
        pack_B<<<(t1 + 255) / 256, 256, 0, stream>>>(W1, Bp1, Kt1, NT1, H);
        int t2 = Kt2 * NT2 * 32;
        pack_B<<<(t2 + 255) / 256, 256, 0, stream>>>(W2, Bp2, Kt2, NT2, C);
    }
    // ---- 3) GEMM1: xw1 = x @ W1 (bf16 WMMA, f32 acc) ----
    {
        dim3 grid((Mtiles + 7) / 8, NT1 / 4);
        gemm_wmma_bf16<4><<<grid, 256, 0, stream>>>(xBf, Bp1, xw1, Mtiles, Kt1, NT1, H);
    }
    // ---- 4) SpMM1: agg1 = S @ xw1 ----
    {
        long waves = E;
        long blocks = (waves * 32 + 255) / 256;
        spmm_atomic<<<(int)blocks, 256, 0, stream>>>(erow, ecol, ev, xw1, agg1, E, H);
    }
    // ---- 5) mid = relu(agg1 + b1); store f32 -> d_out, bf16 -> midBf ----
    {
        long total = (long)N * H;
        bias_relu_store<<<2048, 256, 0, stream>>>(agg1, b1, outMid, midBf, total, H);
    }
    // ---- 6) GEMM2: xw2 = mid @ W2 ----
    {
        dim3 grid((Mtiles + 7) / 8, 1);
        gemm_wmma_bf16<3><<<grid, 256, 0, stream>>>(midBf, Bp2, xw2, Mtiles, Kt2, NT2, C);
    }
    // ---- 7) SpMM2: agg2 = S @ xw2 ----
    {
        long waves = E;
        long blocks = (waves * 32 + 255) / 256;
        spmm_atomic<<<(int)blocks, 256, 0, stream>>>(erow, ecol, ev, xw2, agg2, E, C);
    }
    // ---- 8) out = agg2 + b2 ----
    {
        long total = (long)N * C;
        bias_store<<<1024, 256, 0, stream>>>(agg2, b2, outFin, total, C);
    }
}
